// StaticSparseAttention_33380485824625
// MI455X (gfx1250) — matrix operand order repaired and checked by reading, compiled, not run on hardware
//
#include <hip/hip_runtime.h>

// ---------------- problem constants ----------------
#define Ecfg   768
#define Hh     12
#define Dd     64
#define SEQ    2049
#define LQ     2048           // patch tokens
#define Bb     2
#define Mrows  (Bb*SEQ)       // 4098
#define N_QKV  (3*Ecfg)       // 2304
#define WSZ    16             // half window
#define SPAN   48             // 16-query block key span
#define NKEY   49             // cls + 48 window slots

typedef unsigned short u16;
typedef unsigned int   u32;
typedef __attribute__((ext_vector_type(16))) __bf16 v16bf;
typedef __attribute__((ext_vector_type(8)))  float  v8f;
typedef __attribute__((ext_vector_type(4)))  int    i32x4;

union Frag16 { v16bf v; uint4 q[2]; u16 u[16]; };

__device__ __forceinline__ u16 f2bf(float f) {
    u32 u = __float_as_uint(f);
    u32 r = 0x7FFFu + ((u >> 16) & 1u);
    return (u16)((u + r) >> 16);
}
__device__ __forceinline__ float bf2f(u16 h) {
    return __uint_as_float(((u32)h) << 16);
}

#if defined(__gfx1250__) && __has_builtin(__builtin_amdgcn_global_load_async_to_lds_b128)
#define HAVE_ASYNC_LDS 1
#endif

// ---------------- fp32 -> bf16 convert ----------------
__global__ void cvt_bf16_kernel(const float* __restrict__ src, u16* __restrict__ dst, int n) {
    int i = blockIdx.x * 256 + threadIdx.x;
    if (i < n) dst[i] = f2bf(src[i]);
}

// ---------------- bf16 WMMA GEMM: C[M][N] = A[M][K] * B[N][K]^T + bias ----------------
// 128 threads = 4 waves; each wave owns a 32M x 64N tile (8 accumulators),
// block tile 128M x 64N. Register double-buffered K loop (step 32, unroll 2)
// so global_load_b128 clauses overlap WMMA issue instead of s_wait_loadcnt 0
// before every WMMA. Requires K % 64 == 0.
__global__ __launch_bounds__(128) void gemm_bf16_kernel(
    const u16* __restrict__ A, const u16* __restrict__ Bw,
    const float* __restrict__ bias, float* __restrict__ C,
    int M, int N, int K)
{
    const int wv   = threadIdx.x >> 5;
    const int lane = threadIdx.x & 31;
    const int ll   = lane & 15;
    const int lh   = lane >> 4;
    const int m0   = blockIdx.x * 128 + wv * 32;
    const int n0   = blockIdx.y * 64;

    int r0 = m0 + ll;       if (r0 >= M) r0 = M - 1;   // clamp loads; stores guarded
    int r1 = m0 + 16 + ll;  if (r1 >= M) r1 = M - 1;
    const u16* aOff0 = A + (size_t)r0 * K + lh * 8;    // lanes>=16 start at K+8
    const u16* aOff1 = A + (size_t)r1 * K + lh * 8;
    const u16* bCol[4];
#pragma unroll
    for (int t = 0; t < 4; ++t)
        bCol[t] = Bw + (size_t)(n0 + t * 16 + ll) * K + lh * 8;

    v8f acc[8];
#pragma unroll
    for (int i = 0; i < 8; ++i) acc[i] = (v8f){0.f,0.f,0.f,0.f,0.f,0.f,0.f,0.f};

    Frag16 A0[2], B0[4], A1[2], B1[4];

#define LOADSET(Af, Bf, kk) do {                                          \
        const int _k = (kk);                                              \
        Af[0].q[0] = *(const uint4*)(aOff0 + _k);                         \
        Af[0].q[1] = *(const uint4*)(aOff0 + _k + 16);                    \
        Af[1].q[0] = *(const uint4*)(aOff1 + _k);                         \
        Af[1].q[1] = *(const uint4*)(aOff1 + _k + 16);                    \
        _Pragma("unroll")                                                 \
        for (int _t = 0; _t < 4; ++_t) {                                  \
            Bf[_t].q[0] = *(const uint4*)(bCol[_t] + _k);                 \
            Bf[_t].q[1] = *(const uint4*)(bCol[_t] + _k + 16);            \
        }                                                                 \
    } while (0)

#define WMMASET(Af, Bf) do {                                              \
        _Pragma("unroll")                                                 \
        for (int _t = 0; _t < 4; ++_t) {                                  \
            acc[_t]     = __builtin_amdgcn_wmma_f32_16x16x32_bf16(        \
                false, Af[0].v, false, Bf[_t].v, (short)0, acc[_t], false, false);   \
            acc[4 + _t] = __builtin_amdgcn_wmma_f32_16x16x32_bf16(        \
                false, Af[1].v, false, Bf[_t].v, (short)0, acc[4 + _t], false, false); \
        }                                                                 \
    } while (0)

    LOADSET(A0, B0, 0);
    for (int k0 = 0; k0 < K; k0 += 64) {
        LOADSET(A1, B1, k0 + 32);     // prefetch while set0 computes
        WMMASET(A0, B0);
        LOADSET(A0, B0, k0 + 64);     // final-iter overrun lands in next ws region
        WMMASET(A1, B1);
    }
#undef LOADSET
#undef WMMASET

#pragma unroll
    for (int mi = 0; mi < 2; ++mi)
#pragma unroll
        for (int t = 0; t < 4; ++t) {
            const int col = n0 + t * 16 + ll;
            const float bs = bias ? bias[col] : 0.f;
#pragma unroll
            for (int vi = 0; vi < 8; ++vi) {
                const int row = m0 + mi * 16 + vi + lh * 8;
                if (row < M) C[(size_t)row * N + col] = acc[mi * 4 + t][vi] + bs;
            }
        }
}

// ---------------- RoPE + head split: qkv fp32 -> qb/kb/vb bf16 [B][H][S][D] ----------------
__global__ void rope_split_kernel(const float* __restrict__ qkv, const float* __restrict__ coords,
                                  u16* __restrict__ qb, u16* __restrict__ kb, u16* __restrict__ vb)
{
    const long long total = (long long)Bb * SEQ * Hh * Dd;
    long long idx = (long long)blockIdx.x * 256 + threadIdx.x;
    if (idx >= total) return;
    const int d = (int)(idx % Dd);
    const int h = (int)((idx / Dd) % Hh);
    const int s = (int)((idx / ((long long)Dd * Hh)) % SEQ);
    const int b = (int)(idx / ((long long)Dd * Hh * SEQ));

    const float* row = qkv + (size_t)(b * SEQ + s) * N_QKV;
    const int base  = h * Dd + d;
    const int baseP = h * Dd + (d ^ 1);
    float qv = row[base];
    float kv = row[Ecfg + base];
    float vv = row[2 * Ecfg + base];
    float qo = qv, ko = kv;
    if (s > 0) {
        const float* cd = coords + ((size_t)b * LQ + (s - 1)) * 2;
        const float coord = ((d < 32) ? cd[0] : cd[1]) * 1e-5f;
        const int fi = (d & 31) >> 1;                      // freq index 0..15
        const float invf = __expf(-(float)fi * 0.57564627324851148f); // ln(10000)/16
        const float f = coord * invf;
        float sn, cs;
        __sincosf(f, &sn, &cs);
        const float qp_ = row[baseP];
        const float kp_ = row[Ecfg + baseP];
        const float rq = (d & 1) ? qp_ : -qp_;
        const float rk = (d & 1) ? kp_ : -kp_;
        qo = qv * cs + rq * sn;
        ko = kv * cs + rk * sn;
    }
    const size_t o = (((size_t)(b * Hh + h)) * SEQ + s) * Dd + d;
    qb[o] = f2bf(qo); kb[o] = f2bf(ko); vb[o] = f2bf(vv);
}

// ---------------- CLS-token global attention (one block per (b,h)) ----------------
__global__ __launch_bounds__(256) void cls_attn_kernel(
    const u16* __restrict__ qb, const u16* __restrict__ kb, const u16* __restrict__ vb,
    u16* __restrict__ attn_out)
{
    __shared__ float sc[SEQ];
    __shared__ float red[256];
    __shared__ float qc[Dd];
    __shared__ float partial[4][Dd];
    const int bh = blockIdx.x;
    const int b = bh / Hh, h = bh % Hh;
    const int t = threadIdx.x;
    const u16* qp    = qb + (size_t)bh * SEQ * Dd;   // s = 0 row
    const u16* kbase = kb + (size_t)bh * SEQ * Dd;
    const u16* vbase = vb + (size_t)bh * SEQ * Dd;

    if (t < Dd) qc[t] = bf2f(qp[t]);
    __syncthreads();

    float lmax = -3.0e38f;
    for (int s = t; s < SEQ; s += 256) {
        const u16* kr = kbase + (size_t)s * Dd;
        float dot = 0.f;
        for (int i = 0; i < Dd; ++i) dot += qc[i] * bf2f(kr[i]);
        dot *= 0.125f;
        sc[s] = dot;
        if (dot > lmax) lmax = dot;
    }
    red[t] = lmax; __syncthreads();
    for (int o = 128; o > 0; o >>= 1) { if (t < o && red[t + o] > red[t]) red[t] = red[t + o]; __syncthreads(); }
    const float mx = red[0]; __syncthreads();

    float lsum = 0.f;
    for (int s = t; s < SEQ; s += 256) { float p = __expf(sc[s] - mx); sc[s] = p; lsum += p; }
    red[t] = lsum; __syncthreads();
    for (int o = 128; o > 0; o >>= 1) { if (t < o) red[t] += red[t + o]; __syncthreads(); }
    const float inv = 1.f / red[0];
    __syncthreads();

    const int d = t & 63, pi = t >> 6;       // 4 key chunks x 64 dims
    float a = 0.f;
    for (int s = pi; s < SEQ; s += 4) a += sc[s] * bf2f(vbase[(size_t)s * Dd + d]);
    partial[pi][d] = a; __syncthreads();
    if (t < Dd) {
        const float o = (partial[0][t] + partial[1][t] + partial[2][t] + partial[3][t]) * inv;
        attn_out[(size_t)(b * SEQ) * Ecfg + h * Dd + t] = f2bf(o);
    }
}

// ---------------- windowed attention, WMMA QK^T (one wave = 16 queries) ----------------
__global__ __launch_bounds__(128) void win_attn_kernel(
    const u16* __restrict__ qb, const u16* __restrict__ kb, const u16* __restrict__ vb,
    u16* __restrict__ attn_out)
{
    __shared__ float sS[4][16][52];            // cls + 48 window scores / probs
    __shared__ __align__(16) u16 sV[4][NKEY][Dd];
    const int wv   = threadIdx.x >> 5;
    const int lane = threadIdx.x & 31;
    const int ll   = lane & 15;
    const int lh   = lane >> 4;
    const int bh   = blockIdx.x;               // 0..23
    const int b    = bh / Hh, h = bh % Hh;
    const int l0   = (blockIdx.y * 4 + wv) * 16;
    const u16* qbase = qb + (size_t)bh * SEQ * Dd;
    const u16* kbase = kb + (size_t)bh * SEQ * Dd;
    const u16* vbase = vb + (size_t)bh * SEQ * Dd;

    // stage V (cls + window) into LDS, 16B chunks per lane.
    // Prefer CDNA5 async global->LDS DMA (ASYNCcnt) when the builtin exists.
    for (int it = lane; it < NKEY * 8; it += 32) {
        const int r = it >> 3, ck = it & 7;
        int sidx = 0;
        if (r > 0) {
            int kpos = l0 - WSZ + (r - 1);
            if (kpos < 0) kpos = 0;
            if (kpos >= LQ) kpos = LQ - 1;
            sidx = kpos + 1;
        }
        const u16* gsrc = vbase + (size_t)sidx * Dd + ck * 8;
        u16* ldst = &sV[wv][r][ck * 8];
#ifdef HAVE_ASYNC_LDS
        // param0: int4 in AS1 (global); param1: int4 in AS3 (LDS).
        // Global side: inttoptr is safe (global == generic address values).
        // LDS side: must be a true addrspacecast (strips generic aperture bits).
        __builtin_amdgcn_global_load_async_to_lds_b128(
            (__attribute__((address_space(1))) i32x4*)(unsigned long long)gsrc,
            (__attribute__((address_space(3))) i32x4*)(void*)ldst,
            0, 0);
#else
        *(uint4*)ldst = *(const uint4*)gsrc;
#endif
    }

    // Q fragments (two K-tiles over D)
    Frag16 afr[2];
    {
        const u16* qp = qbase + (size_t)(l0 + ll + 1) * Dd;
#pragma unroll
        for (int kt = 0; kt < 2; ++kt) {
            const int dbase = kt * 32 + lh * 8;
            afr[kt].q[0] = *(const uint4*)(qp + dbase);
            afr[kt].q[1] = *(const uint4*)(qp + dbase + 16);
        }
    }

    // scores via WMMA: 3 key tiles x 2 K-tiles
    v8f accS[3];
#pragma unroll
    for (int t = 0; t < 3; ++t) {
        v8f c = (v8f){0.f,0.f,0.f,0.f,0.f,0.f,0.f,0.f};
        int kpos = l0 - WSZ + t * 16 + ll;
        if (kpos < 0) kpos = 0;
        if (kpos >= LQ) kpos = LQ - 1;
        const u16* kr = kbase + (size_t)(kpos + 1) * Dd;
#pragma unroll
        for (int kt = 0; kt < 2; ++kt) {
            Frag16 bfr;
            const u16* kp = kr + kt * 32 + lh * 8;
            bfr.q[0] = *(const uint4*)(kp);
            bfr.q[1] = *(const uint4*)(kp + 16);
            c = __builtin_amdgcn_wmma_f32_16x16x32_bf16(
                false, afr[kt].v, false, bfr.v, (short)0, c, false, false);
        }
        accS[t] = c;
    }

    // cls score (lanes 0-15: one query row each)
    float scls = 0.f;
    if (lane < 16) {
        const u16* qp = qbase + (size_t)(l0 + lane + 1) * Dd;
        for (int d = 0; d < Dd; ++d) scls += bf2f(qp[d]) * bf2f(kbase[d]);
    }

    // dump scaled scores to LDS
#pragma unroll
    for (int t = 0; t < 3; ++t)
#pragma unroll
        for (int vi = 0; vi < 8; ++vi)
            sS[wv][vi + lh * 8][1 + t * 16 + ll] = accS[t][vi] * 0.125f;
    if (lane < 16) sS[wv][lane][0] = scls * 0.125f;

#ifdef HAVE_ASYNC_LDS
#if __has_builtin(__builtin_amdgcn_s_wait_asynccnt)
    __builtin_amdgcn_s_wait_asynccnt(0);
#else
    asm volatile("s_wait_asynccnt 0x0" ::: "memory");
#endif
#endif
    __syncthreads();

    // masked softmax per query row (lanes 0-15)
    if (lane < 16) {
        const int m = lane;
        float mx = sS[wv][m][0];
        for (int j = 0; j < SPAN; ++j) {
            const int kpos = l0 - WSZ + j;
            const bool ok = (j >= m) && (j <= m + 2 * WSZ) && (kpos >= 0) && (kpos < LQ);
            if (ok) { const float s = sS[wv][m][1 + j]; if (s > mx) mx = s; }
        }
        float sum = 0.f;
        { const float p = __expf(sS[wv][m][0] - mx); sum += p; sS[wv][m][0] = p; }
        for (int j = 0; j < SPAN; ++j) {
            const int kpos = l0 - WSZ + j;
            const bool ok = (j >= m) && (j <= m + 2 * WSZ) && (kpos >= 0) && (kpos < LQ);
            const float p = ok ? __expf(sS[wv][m][1 + j] - mx) : 0.f;
            sum += p; sS[wv][m][1 + j] = p;
        }
        const float inv = 1.f / sum;
        for (int j = 0; j < NKEY; ++j) sS[wv][m][j] *= inv;
    }
    __syncthreads();

    // P * V : lane -> (row = ll, 32 dims starting at lh*32)
    float acc[32];
#pragma unroll
    for (int i = 0; i < 32; ++i) acc[i] = 0.f;
    for (int k = 0; k < NKEY; ++k) {
        const float p = sS[wv][ll][k];
        const u16* vr = &sV[wv][k][lh * 32];
#pragma unroll
        for (int c4 = 0; c4 < 4; ++c4) {
            const uint4 dat = *(const uint4*)(vr + c4 * 8);
            const u32 w0 = dat.x, w1 = dat.y, w2 = dat.z, w3 = dat.w;
            acc[c4*8+0] += p * __uint_as_float(w0 << 16);
            acc[c4*8+1] += p * __uint_as_float(w0 & 0xFFFF0000u);
            acc[c4*8+2] += p * __uint_as_float(w1 << 16);
            acc[c4*8+3] += p * __uint_as_float(w1 & 0xFFFF0000u);
            acc[c4*8+4] += p * __uint_as_float(w2 << 16);
            acc[c4*8+5] += p * __uint_as_float(w2 & 0xFFFF0000u);
            acc[c4*8+6] += p * __uint_as_float(w3 << 16);
            acc[c4*8+7] += p * __uint_as_float(w3 & 0xFFFF0000u);
        }
    }
    const int s = l0 + ll + 1;
    u16* orow = attn_out + (size_t)(b * SEQ + s) * Ecfg + h * Dd + lh * 32;
#pragma unroll
    for (int i = 0; i < 32; i += 2)
        *(u32*)(orow + i) = (u32)f2bf(acc[i]) | ((u32)f2bf(acc[i + 1]) << 16);
}

// ---------------- host launcher ----------------
extern "C" void kernel_launch(void* const* d_in, const int* in_sizes, int n_in,
                              void* d_out, int out_size, void* d_ws, size_t ws_size,
                              hipStream_t stream)
{
    (void)in_sizes; (void)n_in; (void)out_size; (void)ws_size;
    const float* x      = (const float*)d_in[0];
    const float* coords = (const float*)d_in[1];
    const float* qkv_w  = (const float*)d_in[2];
    const float* qkv_b  = (const float*)d_in[3];
    const float* out_w  = (const float*)d_in[4];
    const float* out_b  = (const float*)d_in[5];
    float* out = (float*)d_out;

    char* ws = (char*)d_ws;
    size_t off = 0;
    auto carve = [&](size_t bytes) -> char* {
        char* p = ws + off;
        off += (bytes + 255) & ~(size_t)255;
        return p;
    };
    // Order matters: every buffer that GEMM reads with the K-prefetch overrun
    // (xb, wqkvb, owb, attn) is followed by another region.
    u16*   xb    = (u16*)carve((size_t)Mrows * Ecfg * 2);          // 6.3 MB
    u16*   wqkvb = (u16*)carve((size_t)N_QKV * Ecfg * 2);          // 3.5 MB
    u16*   owb   = (u16*)carve((size_t)Ecfg * Ecfg * 2);           // 1.2 MB
    u16*   qb    = (u16*)carve((size_t)Bb * Hh * SEQ * Dd * 2);    // 6.3 MB
    u16*   kbuf  = (u16*)carve((size_t)Bb * Hh * SEQ * Dd * 2);    // 6.3 MB
    u16*   vbuf  = (u16*)carve((size_t)Bb * Hh * SEQ * Dd * 2);    // 6.3 MB
    u16*   attn  = (u16*)carve((size_t)Mrows * Ecfg * 2);          // 6.3 MB
    float* qkvf  = (float*)carve((size_t)Mrows * N_QKV * 4);       // 37.8 MB
    (void)carve(4096);                                             // safety pad

    {   // fp32 -> bf16
        int n = Mrows * Ecfg;
        cvt_bf16_kernel<<<(n + 255) / 256, 256, 0, stream>>>(x, xb, n);
        n = N_QKV * Ecfg;
        cvt_bf16_kernel<<<(n + 255) / 256, 256, 0, stream>>>(qkv_w, wqkvb, n);
        n = Ecfg * Ecfg;
        cvt_bf16_kernel<<<(n + 255) / 256, 256, 0, stream>>>(out_w, owb, n);
    }

    {   // QKV projection: (4098 x 768) * (2304 x 768)^T
        dim3 g((Mrows + 127) / 128, N_QKV / 64);
        gemm_bf16_kernel<<<g, 128, 0, stream>>>(xb, wqkvb, qkv_b, qkvf, Mrows, N_QKV, Ecfg);
    }

    {   // RoPE + head split
        long long n = (long long)Bb * SEQ * Hh * Dd;
        rope_split_kernel<<<(unsigned)((n + 255) / 256), 256, 0, stream>>>(qkvf, coords, qb, kbuf, vbuf);
    }

    cls_attn_kernel<<<Bb * Hh, 256, 0, stream>>>(qb, kbuf, vbuf, attn);

    {   // windowed attention: 24 (b,h) x 32 query-block groups, 4 waves/block
        dim3 g(Bb * Hh, LQ / 64);
        win_attn_kernel<<<g, 128, 0, stream>>>(qb, kbuf, vbuf, attn);
    }

    {   // output projection: (4098 x 768) * (768 x 768)^T -> fp32 out
        dim3 g((Mrows + 127) / 128, Ecfg / 64);
        gemm_bf16_kernel<<<g, 128, 0, stream>>>(attn, owb, out_b, out, Mrows, Ecfg, Ecfg);
    }
}
